// LSTM2_35433480192803
// MI455X (gfx1250) — matrix operand order repaired and checked by reading, not compiled, not run on hardware
//
#include <hip/hip_runtime.h>
#include <stdint.h>

typedef __bf16 v16bf __attribute__((ext_vector_type(16)));
typedef __bf16 v8bf  __attribute__((ext_vector_type(8)));
typedef float  v8f   __attribute__((ext_vector_type(8)));
typedef unsigned int v4u __attribute__((ext_vector_type(4)));
typedef int v8i __attribute__((ext_vector_type(8)));
typedef int v4i __attribute__((ext_vector_type(4)));

// ---------------- compile-environment probes (printed to stderr, even on success)
#if defined(__gfx1250__)
#if __has_builtin(__builtin_amdgcn_tensor_load_to_lds)
#pragma message("PROBE: HAS __builtin_amdgcn_tensor_load_to_lds")
#else
#pragma message("PROBE: no __builtin_amdgcn_tensor_load_to_lds")
#endif
#if __has_include(<hip/amd_detail/amd_gfx1250_TDM.h>)
#pragma message("PROBE: HAS header amd_gfx1250_TDM.h (6-arg toolchain)")
#else
#pragma message("PROBE: no header amd_gfx1250_TDM.h (5-arg toolchain likely)")
#endif
#if __has_builtin(__builtin_amdgcn_global_load_async_to_lds_b128)
#pragma message("PROBE: HAS __builtin_amdgcn_global_load_async_to_lds_b128")
#else
#pragma message("PROBE: no __builtin_amdgcn_global_load_async_to_lds_b128")
#endif
#if __has_builtin(__builtin_amdgcn_s_wait_asynccnt)
#pragma message("PROBE: HAS __builtin_amdgcn_s_wait_asynccnt")
#else
#pragma message("PROBE: no __builtin_amdgcn_s_wait_asynccnt")
#endif
#if __has_builtin(__builtin_amdgcn_s_wait_tensorcnt)
#pragma message("PROBE: HAS __builtin_amdgcn_s_wait_tensorcnt")
#else
#pragma message("PROBE: no __builtin_amdgcn_s_wait_tensorcnt")
#endif
#if __has_builtin(__builtin_amdgcn_sched_group_barrier)
#pragma message("PROBE: HAS __builtin_amdgcn_sched_group_barrier")
#else
#pragma message("PROBE: no __builtin_amdgcn_sched_group_barrier")
#endif
#endif

// TDM via inline asm: portable across both toolchains (bypasses clang builtin arity skew)
#if defined(__gfx1250__)
#define TDM_ASM 1
#else
#define TDM_ASM 0
#endif

__device__ __forceinline__ unsigned short f2bf(float f) {
    uint32_t u = __float_as_uint(f);
    u += 0x7FFFu + ((u >> 16) & 1u);  // round-to-nearest-even
    return (unsigned short)(u >> 16);
}

// LDS generic pointer: low 32 bits are the wave-relative LDS byte offset (ISA §10.2 aperture rules)
__device__ __forceinline__ unsigned int lds_off(const void* p) {
    return (unsigned int)(uintptr_t)p;
}

#if TDM_ASM
// One TDM descriptor-driven tile load: 2-D bf16 tensor, tile = row_elems x rows,
// LDS padding: after every 16 DWORDs (64B = 32 bf16) insert 8 DWORDs (32B = 16 bf16)
// -> LDS row stride = 48 elements, matching the compute-side fragment indexing.
__device__ __forceinline__ void tdm_load_tile_bf16(const unsigned short* gsrc,
                                                   unsigned int lds_byte_off,
                                                   int rows, int row_elems,
                                                   int tensor_d0, int tensor_d1,
                                                   int row_stride_elems) {
    unsigned long long ga = (unsigned long long)(uintptr_t)gsrc;
    v4u g0;
    g0[0] = 1u;                                                // count=1, user desc
    g0[1] = lds_byte_off;                                      // lds_addr
    g0[2] = (unsigned)(ga & 0xFFFFFFFFu);                      // global_addr[31:0]
    g0[3] = (unsigned)((ga >> 32) & 0x1FFFFFFu) | (2u << 30);  // global_addr[56:32], type=2
    v8i g1;
    g1[0] = (int)((1u << 16)    // data_size = 2 bytes
                | (1u << 20)    // pad_enable
                | (3u << 22)    // pad_interval: 16 DWORDs
                | (7u << 25));  // pad_amount: 8 DWORDs
    g1[1] = (int)(((unsigned)tensor_d0 & 0xFFFFu) << 16);                  // tensor_dim0[15:0]
    g1[2] = (int)((((unsigned)tensor_d0 >> 16) & 0xFFFFu) |
                  (((unsigned)tensor_d1 & 0xFFFFu) << 16));                // d0[31:16], d1[15:0]
    g1[3] = (int)((((unsigned)tensor_d1 >> 16) & 0xFFFFu) |
                  ((unsigned)row_elems << 16));                            // d1[31:16], tile_dim0
    g1[4] = (int)(unsigned)rows;                                           // tile_dim1 (tile_dim2=0)
    g1[5] = (int)(unsigned)row_stride_elems;                               // tensor_dim0_stride[31:0]
    g1[6] = 0;
    g1[7] = 0;
    v4i g2;
    g2[0] = 0; g2[1] = 0; g2[2] = 0; g2[3] = 0;
    v4i g3 = g2;
    asm volatile("tensor_load_to_lds %0, %1, %2, %3"
                 :
                 : "s"(g0), "s"(g1), "s"(g2), "s"(g3)
                 : "memory");
}

__device__ __forceinline__ void tdm_wait0() {
#if __has_builtin(__builtin_amdgcn_s_wait_tensorcnt)
    __builtin_amdgcn_s_wait_tensorcnt(0);
#else
    asm volatile("s_wait_tensorcnt 0x0" ::: "memory");
#endif
}
#endif  // TDM_ASM

// ---------------------------------------------------------------- converters
__global__ void k_f32_to_bf16(const float* __restrict__ src,
                              unsigned short* __restrict__ dst, int n) {
    int i = blockIdx.x * 256 + threadIdx.x;
    if (i < n) dst[i] = f2bf(src[i]);
}

// ---------------------------------------------------------------- layer 1 (K=8, tiny)
__global__ void k_layer1(const float* __restrict__ x, const float* __restrict__ W1,
                         const float* __restrict__ b1, unsigned short* __restrict__ h1) {
    int idx = blockIdx.x * 256 + threadIdx.x;  // B*128 threads
    int b = idx >> 7, j = idx & 127;
    const float* xr = x + b * 8;
    const float* wr = W1 + j * 8;
    float s = b1[j];
#pragma unroll
    for (int k = 0; k < 8; ++k) s = fmaf(xr[k], wr[k], s);
    s = s > 0.f ? s : 0.2f * s;
    h1[idx] = f2bf(s);
}

// ---------------------------------------------------------------- WMMA GEMM (NT)
// C[m][n] = act( sum_k A[m][k] * Bw[n][k] + bias )
//   A_COL=false: A is MxK row-major (bf16).  A_COL=true: A stored as KxM (bf16).
//   Bw: NxK row-major (bf16).  BIAS_M: bias indexed by m, else by n.
template <bool A_COL, bool BIAS_M, bool LRELU, bool OUT_BF16>
__global__ __launch_bounds__(256)
void k_gemm_nt(const unsigned short* __restrict__ A,
               const unsigned short* __restrict__ Bw,
               const float* __restrict__ bias,
               void* __restrict__ Cout, int M, int N, int K) {
    constexpr int LDSS = 48;  // padded row stride (elements): 32 data + 16 pad
    __shared__ __align__(16) unsigned short Ash[2][128 * LDSS];
    __shared__ __align__(16) unsigned short Bsh[2][128 * LDSS];

    const int tid  = threadIdx.x;
    const int m0   = blockIdx.y * 128;
    const int n0   = blockIdx.x * 128;
    const int w    = tid >> 5;  // wave 0..7 -> rows [16w,16w+16)
    const int lane = tid & 31;
    const int lm   = lane & 15;
    const int hi   = lane >> 4;

    v8f acc[8];
#pragma unroll
    for (int i = 0; i < 8; ++i)
#pragma unroll
        for (int j = 0; j < 8; ++j) acc[i][j] = 0.f;

    // manual staging (fallback, and always for the transposed-A source)
    auto stage_manual_row = [&](const unsigned short* src, unsigned short* dstLds,
                                int row0, int ld, int k0) {
        const int r  = tid >> 1;
        const int kh = (tid & 1) * 16;
        const uint4* g = (const uint4*)(src + (size_t)(row0 + r) * ld + k0 + kh);
        uint4 a0 = g[0], a1 = g[1];
        uint4* s = (uint4*)&dstLds[r * LDSS + kh];
        s[0] = a0; s[1] = a1;
    };
    auto stage_manual_Acol = [&](int k0, int buf) {  // A stored KxM; transpose into [m][k]
        const int kk   = tid >> 3;        // 0..31
        const int mseg = (tid & 7) * 16;  // 0..112
        const uint4* g = (const uint4*)(A + (size_t)(k0 + kk) * M + m0 + mseg);
        uint4 a0 = g[0], a1 = g[1];
        const unsigned short* e0 = (const unsigned short*)&a0;
        const unsigned short* e1 = (const unsigned short*)&a1;
#pragma unroll
        for (int j = 0; j < 8; ++j) Ash[buf][(mseg + j) * LDSS + kk] = e0[j];
#pragma unroll
        for (int j = 0; j < 8; ++j) Ash[buf][(mseg + 8 + j) * LDSS + kk] = e1[j];
    };

    auto stage_tile = [&](int k0, int buf) {
#if TDM_ASM
        if (tid < 32) {  // one wave issues the DMA descriptors
            if (!A_COL)
                tdm_load_tile_bf16(A + (size_t)m0 * K + k0, lds_off(&Ash[buf][0]),
                                   128, 32, K, M, K);
            tdm_load_tile_bf16(Bw + (size_t)n0 * K + k0, lds_off(&Bsh[buf][0]),
                               128, 32, K, N, K);
        }
        if (A_COL) stage_manual_Acol(k0, buf);
#else
        if (!A_COL) stage_manual_row(A, &Ash[buf][0], m0, K, k0);
        else        stage_manual_Acol(k0, buf);
        stage_manual_row(Bw, &Bsh[buf][0], n0, K, k0);
#endif
    };

    const int nIter = K / 32;
    int buf = 0;
    stage_tile(0, 0);

    for (int it = 0; it < nIter; ++it) {
#if TDM_ASM
        if (tid < 32) tdm_wait0();  // tile `it` landed in LDS
#endif
        __syncthreads();
        if (it + 1 < nIter) stage_tile((it + 1) * 32, buf ^ 1);  // overlap DMA with WMMA

        const unsigned short* As = &Ash[buf][0];
        const unsigned short* Bs = &Bsh[buf][0];
        // A fragment (16x32): lanes 0-15 -> K 0..7,16..23 ; lanes 16-31 -> K 8..15,24..31
        const int kb = hi ? 8 : 0;
        v8bf alo = *(const v8bf*)&As[(w * 16 + lm) * LDSS + kb];
        v8bf ahi = *(const v8bf*)&As[(w * 16 + lm) * LDSS + 16 + kb];
        v16bf afrag = __builtin_shufflevector(alo, ahi, 0, 1, 2, 3, 4, 5, 6, 7,
                                              8, 9, 10, 11, 12, 13, 14, 15);
        // prefetch ALL B fragments, then issue the 8 WMMAs back-to-back
        const int kbb = hi ? 8 : 0;
        v16bf bfrag[8];
#pragma unroll
        for (int nt = 0; nt < 8; ++nt) {
            v8bf blo = *(const v8bf*)&Bs[(nt * 16 + lm) * LDSS + kbb];
            v8bf bhi = *(const v8bf*)&Bs[(nt * 16 + lm) * LDSS + 16 + kbb];
            bfrag[nt] = __builtin_shufflevector(blo, bhi, 0, 1, 2, 3, 4, 5, 6, 7,
                                                8, 9, 10, 11, 12, 13, 14, 15);
        }
#pragma unroll
        for (int nt = 0; nt < 8; ++nt)
            acc[nt] = __builtin_amdgcn_wmma_f32_16x16x32_bf16(
                false, afrag, false, bfrag[nt], (short)0, acc[nt], false, false);
#if defined(__gfx1250__) && __has_builtin(__builtin_amdgcn_sched_group_barrier)
        __builtin_amdgcn_sched_group_barrier(0x100, 18, 0);  // DS reads first (A+B frags)
        __builtin_amdgcn_sched_group_barrier(0x008, 8, 0);   // then 8 WMMAs back-to-back
#endif
        buf ^= 1;
    }

    // ---- epilogue: bias + activation + store (VGPR v -> M = v + 8*hi) ----
#pragma unroll
    for (int nt = 0; nt < 8; ++nt) {
        const int gn = n0 + nt * 16 + lm;
        const float bn = BIAS_M ? 0.f : bias[gn];
#pragma unroll
        for (int v = 0; v < 8; ++v) {
            const int gm = m0 + w * 16 + 8 * hi + v;
            float c = acc[nt][v] + (BIAS_M ? bias[gm] : bn);
            if (LRELU) c = c > 0.f ? c : 0.2f * c;
            if (OUT_BF16)
                ((unsigned short*)Cout)[(size_t)gm * N + gn] = f2bf(c);
            else
                ((float*)Cout)[(size_t)gm * N + gn] = c;
        }
    }
}

// ---------------------------------------------------------------- LSTM scan (hidden=1)
__global__ void k_lstm(const float* __restrict__ seq_tm, const float* __restrict__ h0,
                       const float* __restrict__ c0, const float* __restrict__ Wih,
                       const float* __restrict__ Whh, const float* __restrict__ bih,
                       const float* __restrict__ bhh, unsigned short* __restrict__ y_tm,
                       int Bn, int T) {
    const int b = blockIdx.x * 256 + threadIdx.x;
    float h = h0[b], c = c0[b];
    const float wi0 = Wih[0], wi1 = Wih[1], wi2 = Wih[2], wi3 = Wih[3];
    const float wh0 = Whh[0], wh1 = Whh[1], wh2 = Whh[2], wh3 = Whh[3];
    const float bb0 = bih[0] + bhh[0], bb1 = bih[1] + bhh[1];
    const float bb2 = bih[2] + bhh[2], bb3 = bih[3] + bhh[3];
    for (int t = 0; t < T; ++t) {
        const float x = seq_tm[(size_t)t * Bn + b];  // coalesced
        const float gi = fmaf(x, wi0, fmaf(h, wh0, bb0));
        const float gf = fmaf(x, wi1, fmaf(h, wh1, bb1));
        const float gg = fmaf(x, wi2, fmaf(h, wh2, bb2));
        const float go = fmaf(x, wi3, fmaf(h, wh3, bb3));
        const float ii = 1.f / (1.f + __expf(-gi));
        const float ff = 1.f / (1.f + __expf(-gf));
        const float g  = tanhf(gg);
        const float oo = 1.f / (1.f + __expf(-go));
        c = fmaf(ff, c, ii * g);
        h = oo * tanhf(c);
        y_tm[(size_t)t * Bn + b] = f2bf(h);  // coalesced, bf16 for final GEMM
    }
}

// ---------------------------------------------------------------- launcher
extern "C" void kernel_launch(void* const* d_in, const int* in_sizes, int n_in,
                              void* d_out, int out_size, void* d_ws, size_t ws_size,
                              hipStream_t stream) {
    (void)in_sizes; (void)n_in; (void)out_size; (void)ws_size;
    const float* x    = (const float*)d_in[0];
    const float* h0   = (const float*)d_in[1];
    const float* c0   = (const float*)d_in[2];
    const float* W1   = (const float*)d_in[3];
    const float* b1   = (const float*)d_in[4];
    const float* W2   = (const float*)d_in[5];
    const float* b2   = (const float*)d_in[6];
    const float* W3   = (const float*)d_in[7];
    const float* b3   = (const float*)d_in[8];
    const float* W4   = (const float*)d_in[9];
    const float* b4   = (const float*)d_in[10];
    const float* Wih  = (const float*)d_in[11];
    const float* Whh  = (const float*)d_in[12];
    const float* bih  = (const float*)d_in[13];
    const float* bhh  = (const float*)d_in[14];
    const float* Wout = (const float*)d_in[15];
    const float* bout = (const float*)d_in[16];
    float* out = (float*)d_out;

    constexpr int Bn = 4096, T = 2048, H1 = 128, H2 = 256, H3 = 1024;

    // workspace carve-up (≈72 MB)
    char* p = (char*)d_ws;
    auto carve = [&](size_t bytes) -> void* {
        void* r = (void*)p;
        p += (bytes + 255) & ~(size_t)255;
        return r;
    };
    unsigned short* W2b   = (unsigned short*)carve((size_t)H2 * H1 * 2);
    unsigned short* W3b   = (unsigned short*)carve((size_t)H3 * H2 * 2);
    unsigned short* W4b   = (unsigned short*)carve((size_t)T * H3 * 2);
    unsigned short* Woutb = (unsigned short*)carve((size_t)T * T * 2);
    unsigned short* h1b   = (unsigned short*)carve((size_t)Bn * H1 * 2);
    unsigned short* h2b   = (unsigned short*)carve((size_t)Bn * H2 * 2);
    unsigned short* h3b   = (unsigned short*)carve((size_t)Bn * H3 * 2);
    float*          seqT  = (float*)carve((size_t)T * Bn * 4);           // time-major (T,B)
    unsigned short* yT    = (unsigned short*)carve((size_t)T * Bn * 2);  // time-major (T,B)

    // weight conversions
    k_f32_to_bf16<<<(H2 * H1 + 255) / 256, 256, 0, stream>>>(W2, W2b, H2 * H1);
    k_f32_to_bf16<<<(H3 * H2 + 255) / 256, 256, 0, stream>>>(W3, W3b, H3 * H2);
    k_f32_to_bf16<<<(T * H3 + 255) / 256, 256, 0, stream>>>(W4, W4b, T * H3);
    k_f32_to_bf16<<<(T * T + 255) / 256, 256, 0, stream>>>(Wout, Woutb, T * T);

    // layer 1 (scalar, K=8)
    k_layer1<<<(Bn * H1) / 256, 256, 0, stream>>>(x, W1, b1, h1b);

    // h2 = lrelu(h1 @ W2^T + b2) : M=Bn, N=H2, K=H1
    k_gemm_nt<false, false, true, true>
        <<<dim3(H2 / 128, Bn / 128), 256, 0, stream>>>(h1b, W2b, b2, h2b, Bn, H2, H1);
    // h3 = lrelu(h2 @ W3^T + b3) : M=Bn, N=H3, K=H2
    k_gemm_nt<false, false, true, true>
        <<<dim3(H3 / 128, Bn / 128), 256, 0, stream>>>(h2b, W3b, b3, h3b, Bn, H3, H2);
    // seqT[t][b] = W4[t]·h3[b] + b4[t] : A=W4 (M=T,K=H3), Bw=h3 (N=Bn), bias on M, f32 out
    k_gemm_nt<false, true, false, false>
        <<<dim3(Bn / 128, T / 128), 256, 0, stream>>>(W4b, h3b, b4, seqT, T, Bn, H3);

    // sequential LSTM scan over T (coalesced time-major I/O)
    k_lstm<<<Bn / 256, 256, 0, stream>>>(seqT, h0, c0, Wih, Whh, bih, bhh, yT, Bn, T);

    // out[b][t'] = y[b]·Wout[t'] + bout[t'] : A=y stored K-major (yT), Bw=Wout, bias on N
    k_gemm_nt<true, false, false, false>
        <<<dim3(T / 128, Bn / 128), 256, 0, stream>>>(yT, Woutb, bout, out, Bn, T, T);
}
